// RelativeAttention_7825430413375
// MI455X (gfx1250) — matrix operand order repaired and checked by reading, not compiled, not run on hardware
//
#include <hip/hip_runtime.h>
#include <hip/hip_bf16.h>
#include <stdint.h>

// ---------------- problem constants ----------------
#define DMODEL 1024
#define NHEAD  16
#define DHEAD  64
#define SEQL   1024
#define RLEN   2048
#define BATCH  4
#define MROWS  (SEQL * BATCH)   // 4096
#define MROWSR (RLEN * BATCH)   // 8192

#ifndef USE_TDM
#define USE_TDM 1               // Tensor Data Mover staging of K into LDS
#endif

typedef _Float16 v8h  __attribute__((ext_vector_type(8)));
typedef _Float16 v16h __attribute__((ext_vector_type(16)));
typedef float    v8f  __attribute__((ext_vector_type(8)));
typedef unsigned int u32x4 __attribute__((ext_vector_type(4)));
typedef int      i32x4 __attribute__((ext_vector_type(4)));
typedef int      i32x8 __attribute__((ext_vector_type(8)));

// Load a 16-half WMMA fragment as two contiguous 16B chunks.
__device__ __forceinline__ v16h ld_frag(const _Float16* lo, const _Float16* hi) {
    v8h a = *(const v8h*)lo;
    v8h b = *(const v8h*)hi;
    v16h r;
#pragma unroll
    for (int i = 0; i < 8; ++i) { r[i] = a[i]; r[i + 8] = b[i]; }
    return r;
}

__device__ __forceinline__ v8f wmma16x16x32(v16h a, v16h b, v8f c) {
    return __builtin_amdgcn_wmma_f32_16x16x32_f16(false, a, false, b, (short)0, c,
                                                  false, false);
}

#if USE_TDM
// 1-D TDM load: n_elem8 x 8-byte elements, global -> LDS (D# per CDNA5 ISA ch.8).
// clang-23 toolchain: 6-arg builtin (g0, g1, g2, g3, i32x8, cpol).
__device__ __forceinline__ void tdm_load_1d_8B(const void* gsrc, unsigned lds_off,
                                               unsigned n_elem8) {
    unsigned long long ga = (unsigned long long)(uintptr_t)gsrc;
    u32x4 g0;
    g0[0] = 1u;                                              // count=1, user mode
    g0[1] = lds_off;                                         // lds_addr (bytes)
    g0[2] = (unsigned)ga;                                    // global_addr[31:0]
    g0[3] = (unsigned)((ga >> 32) & 0x01FFFFFFu) | (2u << 30); // [56:32] | type=2
    i32x8 g1;
    g1[0] = (int)(3u << 16);                                 // data_size = 8 bytes
    g1[1] = (int)((n_elem8 & 0xFFFFu) << 16);                // tensor_dim0[15:0]
    g1[2] = (int)(((n_elem8 >> 16) & 0xFFFFu) | (1u << 16)); // td0 hi | tensor_dim1=1
    g1[3] = (int)((n_elem8 & 0xFFFFu) << 16);                // tile_dim0
    g1[4] = 0;                                               // tile_dim1/2 unused
    g1[5] = (int)n_elem8;                                    // tensor_dim0_stride lo
    g1[6] = 0;
    g1[7] = 0;
    i32x4 z4; z4[0] = 0; z4[1] = 0; z4[2] = 0; z4[3] = 0;
    i32x8 z8;
#pragma unroll
    for (int i = 0; i < 8; ++i) z8[i] = 0;
    __builtin_amdgcn_tensor_load_to_lds(g0, g1, z4, z4, z8, 0);
    __builtin_amdgcn_s_wait_tensorcnt(0);
}
#endif

// ---------------- conversion kernels ----------------
__global__ void cvt_f32_to_f16(const float* __restrict__ in, _Float16* __restrict__ out,
                               int n) {
    int i = blockIdx.x * blockDim.x + threadIdx.x;
    if (i < n) out[i] = (_Float16)in[i];
}

// wT16[col][k] = w[k][col]  (1024x1024)
__global__ void cvt_f32_to_f16_T(const float* __restrict__ in, _Float16* __restrict__ out) {
    int i = blockIdx.x * blockDim.x + threadIdx.x;   // over 1M
    int col = i >> 10, k = i & 1023;
    out[i] = (_Float16)in[k * DMODEL + col];
}

// ---------------- WMMA projection GEMM ----------------
// C[M x 1024] = A[M x 1024] * B,  BT given as [1024(col)][1024(k)] f16.
// One wave computes a 32x32 tile.
// mode 0: Q  -> out1[b][n][i][d] = val + bias1[col]; out2 = val + bias2[col]
// mode 1: K  -> out1[b][n][j][d] = val          (jlen rows per (b,n))
// mode 2: V  -> out1[b][n][d][j] = val          (transposed store)
// mode 3: O  -> outf[m][col] = val + resid[m][col]   (f32)
__global__ __launch_bounds__(32) void proj_wmma(
    const _Float16* __restrict__ A, const _Float16* __restrict__ BT,
    int mode, int jlen,
    _Float16* __restrict__ out1, _Float16* __restrict__ out2,
    const float* __restrict__ bias1, const float* __restrict__ bias2,
    const float* __restrict__ resid, float* __restrict__ outf) {
    const int lane = threadIdx.x;
    const int g = lane >> 4, n15 = lane & 15;
    const int N0 = blockIdx.x * 32;
    const int M0 = blockIdx.y * 32;

    v8f zf = {};
    v8f c[2][2];
    c[0][0] = zf; c[0][1] = zf; c[1][0] = zf; c[1][1] = zf;

    const _Float16* arow0 = A + (size_t)(M0 + n15) * 1024;
    const _Float16* arow1 = A + (size_t)(M0 + 16 + n15) * 1024;
    const _Float16* brow0 = BT + (size_t)(N0 + n15) * 1024;
    const _Float16* brow1 = BT + (size_t)(N0 + 16 + n15) * 1024;

    for (int k0 = 0; k0 < 1024; k0 += 32) {
        __builtin_prefetch(arow0 + k0 + 256, 0, 1);
        __builtin_prefetch(brow0 + k0 + 256, 0, 1);
        v16h a0 = ld_frag(arow0 + k0 + 8 * g, arow0 + k0 + 16 + 8 * g);
        v16h a1 = ld_frag(arow1 + k0 + 8 * g, arow1 + k0 + 16 + 8 * g);
        v16h b0 = ld_frag(brow0 + k0 + 8 * g, brow0 + k0 + 16 + 8 * g);
        v16h b1 = ld_frag(brow1 + k0 + 8 * g, brow1 + k0 + 16 + 8 * g);
        c[0][0] = wmma16x16x32(a0, b0, c[0][0]);
        c[0][1] = wmma16x16x32(a0, b1, c[0][1]);
        c[1][0] = wmma16x16x32(a1, b0, c[1][0]);
        c[1][1] = wmma16x16x32(a1, b1, c[1][1]);
    }

#pragma unroll
    for (int mt = 0; mt < 2; ++mt)
#pragma unroll
        for (int nt = 0; nt < 2; ++nt)
#pragma unroll
            for (int v = 0; v < 8; ++v) {
                int row = M0 + mt * 16 + v + 8 * g;     // D layout: m = v (+8 upper half)
                int col = N0 + nt * 16 + n15;           // n = lane%16
                float val = c[mt][nt][v];
                if (mode == 0) {
                    int i = row >> 2, b = row & 3, hh = col >> 6, d = col & 63;
                    size_t o = ((size_t)(b * NHEAD + hh) * SEQL + i) * DHEAD + d;
                    out1[o] = (_Float16)(val + bias1[col]);
                    out2[o] = (_Float16)(val + bias2[col]);
                } else if (mode == 1) {
                    int j = row >> 2, b = row & 3, hh = col >> 6, d = col & 63;
                    out1[((size_t)(b * NHEAD + hh) * jlen + j) * DHEAD + d] = (_Float16)val;
                } else if (mode == 2) {
                    int j = row >> 2, b = row & 3, hh = col >> 6, d = col & 63;
                    out1[((size_t)(b * NHEAD + hh) * DHEAD + d) * jlen + j] = (_Float16)val;
                } else {
                    outf[(size_t)row * DMODEL + col] =
                        val + resid[(size_t)row * DMODEL + col];
                }
            }
}

// ---------------- fused flash attention with rel-shift ----------------
// grid (B*NHEAD, SEQ/64), block 128 (4 waves); wave handles one 16-row i-tile.
// K for this (b,n) is staged into LDS in 4x 32KB segments by the Tensor Data
// Mover (wave 0 issues TENSOR_LOAD_TO_LDS, waits TENSORcnt, block barrier).
// logits[i,j] = (q_wb[i].k[j] + q_rb[i].k_r[j + SEQ - i]) * 1/8, online softmax,
// O = P*V, write attn_vec f16 to av[(i*4+b)][n*64+d].
#define KSEG 256   // K rows staged per segment
__global__ __launch_bounds__(128) void flash_relattn(
    const _Float16* __restrict__ qwb, const _Float16* __restrict__ qrb,
    const _Float16* __restrict__ kk,  const _Float16* __restrict__ kr,
    const _Float16* __restrict__ vT,  _Float16* __restrict__ av) {
    const int lane = threadIdx.x & 31;
    const int wid  = threadIdx.x >> 5;
    const int g = lane >> 4, n15 = lane & 15;
    const int bn = blockIdx.x;                         // b*16 + n
    const int b  = bn >> 4, nh = bn & 15;
    const int i0 = (blockIdx.y * 4 + wid) * 16;

    __shared__ __align__(16) _Float16 kbuf[KSEG * DHEAD];   // 32 KB K segment
    __shared__ __align__(32) _Float16 pbuf[4][16 * 32];     // 4 KB prob bounce
    _Float16* pb = pbuf[wid];

    // Q fragments (A operand), K split d:[0,32) and [32,64)
    const _Float16* q0 = qwb + ((size_t)bn * SEQL + i0 + n15) * DHEAD;
    const _Float16* q1 = qrb + ((size_t)bn * SEQL + i0 + n15) * DHEAD;
    v16h awb0 = ld_frag(q0 + 16 * g,      q0 + 8 + 16 * g);
    v16h awb1 = ld_frag(q0 + 32 + 16 * g, q0 + 40 + 16 * g);
    v16h arb0 = ld_frag(q1 + 16 * g,      q1 + 8 + 16 * g);
    v16h arb1 = ld_frag(q1 + 32 + 16 * g, q1 + 40 + 16 * g);

    v8f zf = {};
    v8f O[4]; O[0] = zf; O[1] = zf; O[2] = zf; O[3] = zf;
    float Mx[8], Sx[8];
#pragma unroll
    for (int v = 0; v < 8; ++v) { Mx[v] = -1e30f; Sx[v] = 0.f; }

    for (int seg = 0; seg < SEQL / KSEG; ++seg) {
        __syncthreads();   // previous segment fully consumed before overwrite
#if USE_TDM
        if (wid == 0) {
            tdm_load_1d_8B(kk + ((size_t)bn * SEQL + seg * KSEG) * DHEAD,
                           (unsigned)(uintptr_t)(void*)kbuf,
                           (unsigned)(KSEG * DHEAD * 2 / 8));
        }
#else
        {
            const v8h* src = (const v8h*)(kk + ((size_t)bn * SEQL + seg * KSEG) * DHEAD);
            v8h* dst = (v8h*)kbuf;
            for (int t = threadIdx.x; t < KSEG * DHEAD / 8; t += 128) dst[t] = src[t];
        }
#endif
        __syncthreads();   // publish staged K to all waves

        for (int jc = 0; jc < KSEG / 32; ++jc) {
            const int j0c  = seg * KSEG + jc * 32;
            const int jloc = jc * 32;
            // ---- content scores from LDS-staged K (two 16-col tiles) ----
            v8f s0 = zf, s1 = zf;
            {
                const _Float16* kb = kbuf + (size_t)(jloc + n15) * DHEAD + 16 * g;
                s0 = wmma16x16x32(awb0, ld_frag(kb, kb + 8), s0);
                s0 = wmma16x16x32(awb1, ld_frag(kb + 32, kb + 40), s0);
                const _Float16* kc = kbuf + (size_t)(jloc + 16 + n15) * DHEAD + 16 * g;
                s1 = wmma16x16x32(awb0, ld_frag(kc, kc + 8), s1);
                s1 = wmma16x16x32(awb1, ld_frag(kc + 32, kc + 40), s1);
            }
            // ---- positional scores: 3-tile strip of bd_orig at base A0 ----
            const int A0 = j0c + (SEQL - 16) - i0;      // >= 0, A0+48 <= RLEN
            v8f G[3];
#pragma unroll
            for (int t = 0; t < 3; ++t) {
                const _Float16* rb =
                    kr + ((size_t)bn * RLEN + A0 + 16 * t + n15) * DHEAD + 16 * g;
                v8f acc = zf;
                acc = wmma16x16x32(arb0, ld_frag(rb, rb + 8), acc);
                acc = wmma16x16x32(arb1, ld_frag(rb + 32, rb + 40), acc);
                G[t] = acc;
            }
            // ---- rel-shift extraction: bd[m, n] = strip[m, 16 + n - m] ----
#pragma unroll
            for (int v = 0; v < 8; ++v) {
                int m = v + 8 * g;
                int Ls = (lane & 16) | ((n15 - m) & 15);
                float g0 = __shfl(G[0][v], Ls);
                float g1 = __shfl(G[1][v], Ls);
                float g2 = __shfl(G[2][v], Ls);
                float bd0 = (n15 >= m) ? g1 : g0;   // tile j0c   : strip (G0,G1)
                float bd1 = (n15 >= m) ? g2 : g1;   // tile j0c+16: strip (G1,G2)
                s0[v] = (s0[v] + bd0) * 0.125f;
                s1[v] = (s1[v] + bd1) * 0.125f;
            }
            // ---- online softmax update (rows live across the 16-lane group) ----
#pragma unroll
            for (int v = 0; v < 8; ++v) {
                float tm = fmaxf(s0[v], s1[v]);
#pragma unroll
                for (int msk = 1; msk < 16; msk <<= 1)
                    tm = fmaxf(tm, __shfl_xor(tm, msk));
                float newM = fmaxf(Mx[v], tm);
                float corr = __expf(Mx[v] - newM);
                float p0 = __expf(s0[v] - newM);
                float p1 = __expf(s1[v] - newM);
                float rs = p0 + p1;
#pragma unroll
                for (int msk = 1; msk < 16; msk <<= 1) rs += __shfl_xor(rs, msk);
                Sx[v] = Sx[v] * corr + rs;
                Mx[v] = newM;
                O[0][v] *= corr; O[1][v] *= corr; O[2][v] *= corr; O[3][v] *= corr;
                s0[v] = p0; s1[v] = p1;
            }
            // ---- P: D-layout f32 -> A-layout f16 via LDS bounce ----
            __syncthreads();
#pragma unroll
            for (int v = 0; v < 8; ++v) {
                int m = v + 8 * g;
                pb[m * 32 + n15]      = (_Float16)s0[v];
                pb[m * 32 + 16 + n15] = (_Float16)s1[v];
            }
            __syncthreads();
            v16h pA = ld_frag(pb + n15 * 32 + 16 * g, pb + n15 * 32 + 8 + 16 * g);
            // ---- O += P * V (vT is [b][n][d][j]) ----
#pragma unroll
            for (int t = 0; t < 4; ++t) {
                const _Float16* vb =
                    vT + ((size_t)bn * DHEAD + t * 16 + n15) * SEQL + j0c + 16 * g;
                O[t] = wmma16x16x32(pA, ld_frag(vb, vb + 8), O[t]);
            }
        }
    }

    // ---- normalize and store attn_vec ----
#pragma unroll
    for (int v = 0; v < 8; ++v) {
        float inv = 1.0f / Sx[v];
        int m = v + 8 * g;
        size_t row = (size_t)(i0 + m) * BATCH + b;
#pragma unroll
        for (int t = 0; t < 4; ++t)
            av[row * DMODEL + nh * DHEAD + t * 16 + n15] = (_Float16)(O[t][v] * inv);
    }
}

// ---------------- LayerNorm ----------------
__global__ __launch_bounds__(256) void layernorm_rows(
    const float* __restrict__ x, const float* __restrict__ gamma,
    const float* __restrict__ beta, float* __restrict__ out) {
    const int row = blockIdx.x;
    const float* xr = x + (size_t)row * DMODEL;
    float s = 0.f, s2 = 0.f;
    for (int c = threadIdx.x; c < DMODEL; c += 256) {
        float v = xr[c]; s += v; s2 += v * v;
    }
    __shared__ float sh0[256], sh1[256];
    sh0[threadIdx.x] = s; sh1[threadIdx.x] = s2;
    __syncthreads();
    for (int off = 128; off > 0; off >>= 1) {
        if (threadIdx.x < off) {
            sh0[threadIdx.x] += sh0[threadIdx.x + off];
            sh1[threadIdx.x] += sh1[threadIdx.x + off];
        }
        __syncthreads();
    }
    float mu = sh0[0] * (1.0f / DMODEL);
    float var = sh1[0] * (1.0f / DMODEL) - mu * mu;
    float rstd = rsqrtf(var + 1e-5f);
    for (int c = threadIdx.x; c < DMODEL; c += 256)
        out[(size_t)row * DMODEL + c] = (xr[c] - mu) * rstd * gamma[c] + beta[c];
}

// ---------------- host-side orchestration ----------------
extern "C" void kernel_launch(void* const* d_in, const int* in_sizes, int n_in,
                              void* d_out, int out_size, void* d_ws, size_t ws_size,
                              hipStream_t stream) {
    (void)in_sizes; (void)n_in; (void)out_size; (void)ws_size;
    const float* h        = (const float*)d_in[0];
    const float* r        = (const float*)d_in[1];
    const float* q_w      = (const float*)d_in[2];
    const float* k_w      = (const float*)d_in[3];
    const float* v_w      = (const float*)d_in[4];
    const float* o_w      = (const float*)d_in[5];
    const float* r_w      = (const float*)d_in[6];
    const float* r_r_bias = (const float*)d_in[7];
    const float* r_w_bias = (const float*)d_in[8];
    const float* ln_g     = (const float*)d_in[9];
    const float* ln_b     = (const float*)d_in[10];
    float* out = (float*)d_out;

    char* p = (char*)d_ws;
    auto alloc = [&](size_t bytes) -> char* {
        char* q = p; p += (bytes + 255) & ~(size_t)255; return q;
    };
    _Float16* h16   = (_Float16*)alloc((size_t)MROWS  * DMODEL * 2);
    _Float16* r16   = (_Float16*)alloc((size_t)MROWSR * DMODEL * 2);
    _Float16* qwT   = (_Float16*)alloc((size_t)DMODEL * DMODEL * 2);
    _Float16* kwT   = (_Float16*)alloc((size_t)DMODEL * DMODEL * 2);
    _Float16* vwT   = (_Float16*)alloc((size_t)DMODEL * DMODEL * 2);
    _Float16* rwT   = (_Float16*)alloc((size_t)DMODEL * DMODEL * 2);
    _Float16* ow16  = (_Float16*)alloc((size_t)DMODEL * DMODEL * 2);
    _Float16* qwb16 = (_Float16*)alloc((size_t)MROWS * DMODEL * 2);
    _Float16* qrb16 = (_Float16*)alloc((size_t)MROWS * DMODEL * 2);
    _Float16* k16   = (_Float16*)alloc((size_t)MROWS * DMODEL * 2);
    _Float16* vT16  = (_Float16*)alloc((size_t)MROWS * DMODEL * 2);
    _Float16* kr16  = (_Float16*)alloc((size_t)MROWSR * DMODEL * 2);
    _Float16* av16  = (_Float16*)alloc((size_t)MROWS * DMODEL * 2);
    float*    attn  = (float*)alloc((size_t)MROWS * DMODEL * 4);

    // 1) f32 -> f16 conversions (weights transposed to [col][k])
    {
        int n = MROWS * DMODEL;
        cvt_f32_to_f16<<<(n + 255) / 256, 256, 0, stream>>>(h, h16, n);
        n = MROWSR * DMODEL;
        cvt_f32_to_f16<<<(n + 255) / 256, 256, 0, stream>>>(r, r16, n);
        int nw = DMODEL * DMODEL;
        cvt_f32_to_f16_T<<<nw / 256, 256, 0, stream>>>(q_w, qwT);
        cvt_f32_to_f16_T<<<nw / 256, 256, 0, stream>>>(k_w, kwT);
        cvt_f32_to_f16_T<<<nw / 256, 256, 0, stream>>>(v_w, vwT);
        cvt_f32_to_f16_T<<<nw / 256, 256, 0, stream>>>(r_w, rwT);
        cvt_f32_to_f16<<<nw / 256, 256, 0, stream>>>(o_w, ow16, nw);
    }

    // 2) projections (WMMA)
    dim3 gH(DMODEL / 32, MROWS / 32), gR(DMODEL / 32, MROWSR / 32);
    proj_wmma<<<gH, 32, 0, stream>>>(h16, qwT, 0, SEQL, qwb16, qrb16,
                                     r_w_bias, r_r_bias, nullptr, nullptr);
    proj_wmma<<<gH, 32, 0, stream>>>(h16, kwT, 1, SEQL, k16, nullptr,
                                     nullptr, nullptr, nullptr, nullptr);
    proj_wmma<<<gH, 32, 0, stream>>>(h16, vwT, 2, SEQL, vT16, nullptr,
                                     nullptr, nullptr, nullptr, nullptr);
    proj_wmma<<<gR, 32, 0, stream>>>(r16, rwT, 1, RLEN, kr16, nullptr,
                                     nullptr, nullptr, nullptr, nullptr);

    // 3) fused rel-shift flash attention (WMMA + TDM K staging)
    flash_relattn<<<dim3(BATCH * NHEAD, SEQL / 64), 128, 0, stream>>>(
        qwb16, qrb16, k16, kr16, vT16, av16);

    // 4) output projection + residual (WMMA, f32 out)
    proj_wmma<<<gH, 32, 0, stream>>>(av16, ow16, 3, SEQL, nullptr, nullptr,
                                     nullptr, nullptr, h, attn);

    // 5) LayerNorm -> d_out
    layernorm_rows<<<MROWS, 256, 0, stream>>>(attn, ln_g, ln_b, out);
}
